// ClassCentersEMA_31782757990437
// MI455X (gfx1250) — matrix operand order repaired and checked by reading, not compiled, not run on hardware
//
#include <hip/hip_runtime.h>
#include <hip/hip_bf16.h>

typedef __attribute__((ext_vector_type(16))) _Float16 v16h;
typedef __attribute__((ext_vector_type(8)))  float    v8f;
typedef __attribute__((ext_vector_type(4)))  float    f4v;
typedef __attribute__((ext_vector_type(4)))  unsigned int v4u;

#define MOM   0.9f
#define EPSN  1e-12f
#define MB    64
#define NB    128
#define KB    64
#define SPLITK 8

union HV { v4u q[2]; v16h v; };
union HPK { _Float16 h[2]; unsigned int u; };

// ---------------------------------------------------------------- zero scratch
__global__ void zero_ws_kernel(float* __restrict__ p, int n) {
    int i = blockIdx.x * 256 + threadIdx.x;
    if (i < n) p[i] = 0.0f;
}

// ------------------------------------------------- inv[b] = 1/max(||f_b||,eps)
__global__ void row_inv_norm_kernel(const float* __restrict__ feats,
                                    float* __restrict__ inv, int D) {
    __shared__ float red[256];
    int b = blockIdx.x, t = threadIdx.x;
    const float* row = feats + (size_t)b * D;
    float s = 0.0f;
    for (int i = t; i < D; i += 256) { float v = row[i]; s += v * v; }
    red[t] = s; __syncthreads();
    for (int o = 128; o > 0; o >>= 1) {
        if (t < o) red[t] += red[t + o];
        __syncthreads();
    }
    if (t == 0) inv[b] = 1.0f / fmaxf(sqrtf(red[0]), EPSN);
}

// --------------------------------------------------- counts[c] = sum_b t[b,c]
// Partial sums are integer-valued (multi-hot 0/1) so float atomics are exact
// and order-independent -> deterministic.
__global__ void class_counts_kernel(const float* __restrict__ targets,
                                    float* __restrict__ counts,
                                    int C, int rowsPerBlk) {
    int c  = blockIdx.x * 256 + threadIdx.x;
    int b0 = blockIdx.y * rowsPerBlk;
    float s = 0.0f;
    for (int b = b0; b < b0 + rowsPerBlk; ++b)
        s += targets[(size_t)b * C + c];
    atomicAdd(&counts[c], s);
}

// ------------------------------------------------------------------- the GEMM
// parts[z][c][d] += sum_{k in split z} (targets[k,c]*inv[k]) * feats[k,d]
// A = targets^T scaled (M=C, K=B), B = feats (K=B, N=D).
// Block tile 64x128x64, 8 waves: wm = wid>>1 (4 x 16 rows), wn = wid&1 (2 x 64 cols),
// each wave owns four 16x16 f32 accumulators sharing one A fragment.
__global__ void __launch_bounds__(256)
ema_gemm_kernel(const float* __restrict__ targets,
                const float* __restrict__ feats,
                const float* __restrict__ inv,
                float* __restrict__ parts,
                int B, int C, int D) {
    __shared__ __align__(16) _Float16 Al[MB * KB];   // [m][k], k contiguous
    __shared__ __align__(16) _Float16 Bl[NB * KB];   // [n][k], k contiguous
    unsigned int* Alu = (unsigned int*)Al;           // packed k-pairs
    unsigned int* Blu = (unsigned int*)Bl;

    const int t  = threadIdx.x;
    const int c0 = blockIdx.y * MB;
    const int d0 = blockIdx.x * NB;
    const int kPerSplit = B / SPLITK;
    const int kbase = blockIdx.z * kPerSplit;
    const int kend  = kbase + kPerSplit;
    float* out = parts + (size_t)blockIdx.z * C * D;

    const int wid  = t >> 5, lane = t & 31;
    const int wm   = wid >> 1, wn = wid & 1;
    const int l15  = lane & 15, hi = lane >> 4;

    // staging coordinates: each thread stages column quads (b128 global loads)
    const int cqA = (t & 15) * 4;   // A: 16 quads cover 64 columns
    const int pA0 = t >> 4;         //    k-pair base 0..15
    const int nqB = (t & 31) * 4;   // B: 32 quads cover 128 columns
    const int pB0 = t >> 5;         //    k-pair base 0..7

    v8f acc0 = {0,0,0,0,0,0,0,0};
    v8f acc1 = {0,0,0,0,0,0,0,0};
    v8f acc2 = {0,0,0,0,0,0,0,0};
    v8f acc3 = {0,0,0,0,0,0,0,0};

    for (int k0 = kbase; k0 < kend; k0 += KB) {
        // ---- stage A: 64x64 f16. b128 global loads (2 adjacent K rows of a
        // column quad), scale by inv[k], cvt_pk to f16 pairs, b32 LDS stores.
        #pragma unroll
        for (int i = 0; i < 2; ++i) {
            int p = pA0 + 16 * i;
            int k = k0 + 2 * p;
            f4v r0 = *(const f4v*)&targets[(size_t)k       * C + (c0 + cqA)];
            f4v r1 = *(const f4v*)&targets[(size_t)(k + 1) * C + (c0 + cqA)];
            float i0 = inv[k], i1 = inv[k + 1];
            #pragma unroll
            for (int j = 0; j < 4; ++j) {
                HPK pk;
                pk.h[0] = (_Float16)(r0[j] * i0);
                pk.h[1] = (_Float16)(r1[j] * i1);
                Alu[(cqA + j) * (KB / 2) + p] = pk.u;
            }
        }
        // ---- stage B: 128x64 f16, same scheme (no scaling)
        #pragma unroll
        for (int i = 0; i < 4; ++i) {
            int p = pB0 + 8 * i;
            int k = k0 + 2 * p;
            f4v r0 = *(const f4v*)&feats[(size_t)k       * D + (d0 + nqB)];
            f4v r1 = *(const f4v*)&feats[(size_t)(k + 1) * D + (d0 + nqB)];
            #pragma unroll
            for (int j = 0; j < 4; ++j) {
                HPK pk;
                pk.h[0] = (_Float16)r0[j];
                pk.h[1] = (_Float16)r1[j];
                Blu[(nqB + j) * (KB / 2) + p] = pk.u;
            }
        }
        // prefetch next K-tile while this one computes (global_prefetch_b8)
        if (k0 + KB < kend) {
            __builtin_prefetch(&targets[(size_t)(k0 + KB + (t >> 2)) * C + c0 + (t & 3) * 16], 0, 0);
            __builtin_prefetch(&feats[(size_t)(k0 + KB + (t >> 2)) * D + d0 + (t & 3) * 32], 0, 0);
        }
        __syncthreads();

        const int mrow = wm * 16 + l15;
        #pragma unroll
        for (int kk = 0; kk < KB; kk += 32) {
            // A fragment: lane (m=l15) holds K = [hi*16, hi*16+16) contiguous
            HV a;
            a.q[0] = *(const v4u*)&Al[mrow * KB + kk + hi * 16];
            a.q[1] = *(const v4u*)&Al[mrow * KB + kk + hi * 16 + 8];
            // B fragments: lane (n=l15) holds K = [hi*16, hi*16+16) contiguous
            const int bk = kk + hi * 16;
            HV b0, b1, b2, b3;
            const _Float16* bp0 = &Bl[(wn * 64 +  0 + l15) * KB + bk];
            const _Float16* bp1 = &Bl[(wn * 64 + 16 + l15) * KB + bk];
            const _Float16* bp2 = &Bl[(wn * 64 + 32 + l15) * KB + bk];
            const _Float16* bp3 = &Bl[(wn * 64 + 48 + l15) * KB + bk];
            b0.q[0] = *(const v4u*)(bp0);  b0.q[1] = *(const v4u*)(bp0 + 8);
            b1.q[0] = *(const v4u*)(bp1);  b1.q[1] = *(const v4u*)(bp1 + 8);
            b2.q[0] = *(const v4u*)(bp2);  b2.q[1] = *(const v4u*)(bp2 + 8);
            b3.q[0] = *(const v4u*)(bp3);  b3.q[1] = *(const v4u*)(bp3 + 8);

            acc0 = __builtin_amdgcn_wmma_f32_16x16x32_f16(
                       false, a.v, false, b0.v, (short)0, acc0, false, false);
            acc1 = __builtin_amdgcn_wmma_f32_16x16x32_f16(
                       false, a.v, false, b1.v, (short)0, acc1, false, false);
            acc2 = __builtin_amdgcn_wmma_f32_16x16x32_f16(
                       false, a.v, false, b2.v, (short)0, acc2, false, false);
            acc3 = __builtin_amdgcn_wmma_f32_16x16x32_f16(
                       false, a.v, false, b3.v, (short)0, acc3, false, false);
        }
        __syncthreads();
    }

    // D layout: VGPR r <-> row (r + 8*hi), col = l15
    #pragma unroll
    for (int r = 0; r < 8; ++r) {
        int row = c0 + wm * 16 + r + hi * 8;
        size_t rb = (size_t)row * D + d0 + wn * 64 + l15;
        out[rb     ] = acc0[r];
        out[rb + 16] = acc1[r];
        out[rb + 32] = acc2[r];
        out[rb + 48] = acc3[r];
    }
}

// ------------------------------------------------------- EMA blend + renorm
__global__ void finalize_kernel(const float* __restrict__ parts,
                                const float* __restrict__ counts,
                                const float* __restrict__ centers,
                                const unsigned char* __restrict__ inited,
                                float* __restrict__ out, int C, int D) {
    __shared__ float red[256];
    const int c = blockIdx.x, t = threadIdx.x;
    const float cnt  = counts[c];
    const bool  mask = cnt > 0.0f;
    const bool  newc = (inited[c] == 0) && mask;
    const size_t base = (size_t)c * D;
    const size_t CD   = (size_t)C * D;

    float vals[8];   // supports D up to 2048 with 256 threads (D=1024 -> 4 used)
    const int nIt = D / 256;
    float ss = 0.0f;
    for (int i = 0; i < nIt; ++i) {
        int d = t + i * 256;
        float s = 0.0f;
        #pragma unroll
        for (int sp = 0; sp < SPLITK; ++sp) s += parts[sp * CD + base + d];
        float mean = mask ? s / fmaxf(cnt, 1.0f) : 0.0f;
        float ctr  = centers[base + d];
        float bse  = newc ? mean : ctr;
        float bl   = MOM * bse + (1.0f - MOM) * mean;
        float u    = mask ? bl : ctr;
        vals[i] = u;
        ss += u * u;
    }
    red[t] = ss; __syncthreads();
    for (int o = 128; o > 0; o >>= 1) {
        if (t < o) red[t] += red[t + o];
        __syncthreads();
    }
    const float scale = mask ? 1.0f / fmaxf(sqrtf(red[0]), EPSN) : 1.0f;
    for (int i = 0; i < nIt; ++i)
        out[base + t + i * 256] = vals[i] * scale;
}

// --------------------------------------------------------------------- launch
extern "C" void kernel_launch(void* const* d_in, const int* in_sizes, int n_in,
                              void* d_out, int out_size, void* d_ws, size_t ws_size,
                              hipStream_t stream) {
    (void)n_in; (void)out_size; (void)ws_size;
    const float* feats   = (const float*)d_in[0];   // [B, D]
    const float* targets = (const float*)d_in[1];   // [B, C]
    const float* centers = (const float*)d_in[2];   // [C, D]
    const unsigned char* inited = (const unsigned char*)d_in[3]; // [C] bool
    float* out = (float*)d_out;                     // [C, D]

    const int C = in_sizes[3];          // 512
    const int D = in_sizes[2] / C;      // 1024
    const int B = in_sizes[0] / D;      // 16384

    // workspace layout
    char* ws = (char*)d_ws;
    float* inv    = (float*)ws;                                   // B floats
    float* counts = (float*)(ws + (size_t)B * 4);                 // C floats
    float* parts  = (float*)(ws + (size_t)B * 4 + 4096);          // SPLITK*C*D

    const int nParts = SPLITK * C * D;
    zero_ws_kernel<<<(C + 255) / 256, 256, 0, stream>>>(counts, C);
    zero_ws_kernel<<<(nParts + 255) / 256, 256, 0, stream>>>(parts, nParts);

    row_inv_norm_kernel<<<B, 256, 0, stream>>>(feats, inv, D);

    dim3 gc(C / 256, 32);
    class_counts_kernel<<<gc, 256, 0, stream>>>(targets, counts, C, B / 32);

    dim3 gg(D / NB, C / MB, SPLITK);
    ema_gemm_kernel<<<gg, 256, 0, stream>>>(targets, feats, inv, parts, B, C, D);

    finalize_kernel<<<C, 256, 0, stream>>>(parts, counts, centers, inited, out, C, D);
}
